// MultiHeadSelfAttention_32701880992467
// MI455X (gfx1250) — matrix operand order repaired and checked by reading, not compiled, not run on hardware
//
#include <hip/hip_runtime.h>

// ---------------------------------------------------------------------------
// MHA forward for MI455X (gfx1250, wave32, WMMA).
// B=4, T=2048, D=768, H=12, DH=64.  ~90 GFLOP, all tensors fit in 192MB L2
// => compute-bound on the matrix pipe: everything via v_wmma_f32_16x16x32_bf16,
// with register-level double buffering so WMMAs don't stall on loadcnt.
// ---------------------------------------------------------------------------

#define B_  4
#define T_  2048
#define D_  768
#define H_  12
#define DH_ 64

typedef __attribute__((ext_vector_type(16))) __bf16 v16bf;
typedef __attribute__((ext_vector_type(8)))  __bf16 v8bf;
typedef __attribute__((ext_vector_type(8)))  float  v8f;

static __device__ __forceinline__ unsigned short f2bf(float f) {
  union { float f; unsigned int u; } v; v.f = f;
  unsigned int r = v.u + 0x7FFFu + ((v.u >> 16) & 1u);   // round-to-nearest-even
  return (unsigned short)(r >> 16);
}

// Build a 16-half fragment from two 16-byte chunks.
static __device__ __forceinline__ v16bf ldfrag(const __bf16* p0, const __bf16* p1) {
  v8bf lo = *(const v8bf*)p0;
  v8bf hi = *(const v8bf*)p1;
  return __builtin_shufflevector(lo, hi, 0,1,2,3,4,5,6,7,8,9,10,11,12,13,14,15);
}

static __device__ __forceinline__ v8f wmma_bf16(v16bf a, v16bf b, v8f c) {
  return __builtin_amdgcn_wmma_f32_16x16x32_bf16(false, a, false, b, (short)0, c,
                                                 false, false);
}

// ---------------------------------------------------------------------------
// Stage 1: f32 -> bf16 cast (vectorized; all sizes divisible by 4)
// ---------------------------------------------------------------------------
__global__ __launch_bounds__(256) void cast_f32_bf16(const float* __restrict__ in,
                                                     unsigned short* __restrict__ out,
                                                     int n) {
  int i = (blockIdx.x * 256 + threadIdx.x) * 4;
  if (i < n) {
    float4 f = *(const float4*)(in + i);
    ushort4 o;
    o.x = f2bf(f.x); o.y = f2bf(f.y); o.z = f2bf(f.z); o.w = f2bf(f.w);
    *(ushort4*)(out + i) = o;
  }
}

// ---------------------------------------------------------------------------
// Stage 2/5: C[M,N] = A[M,K] @ W[N,K]^T via WMMA bf16, software-pipelined.
// Block = 256 thr (8 waves = 2 m-waves x 4 n-waves), block tile 64x256.
// Wave tile 32x64: 2 A-frags x 4 B-frags -> 8 WMMAs per 6 fragment loads.
// k-loop double-buffers fragments in registers: next step's loads issue
// before current step's WMMAs so s_wait_loadcnt overlaps with matrix work.
// A-frag (16x32): lane&15 = row; lane<16 -> K 0..15, else 16..31 (contiguous)
// B-frag (32x16): lane&15 = col; lane<16 -> K 0..15 else 16..31 (contiguous:
//                 W is (N,K) row-major = column-major KxN)
// C (f32 16x16):  vgpr r -> row r + 8*(lane>=16); col = lane&15
// mode 0: bf16 out [B,H,T,DH] (Q: scale=1/8, K: scale=1)
// mode 1: bf16 out [B,H,DH,T] (V transposed -> contiguous PV B-frags)
// mode 2: f32  out row-major [M,N] (final projection -> d_out)
// ---------------------------------------------------------------------------
__global__ __launch_bounds__(256) void gemm_bf16(const __bf16* __restrict__ A,
                                                 const __bf16* __restrict__ W,
                                                 void* __restrict__ out,
                                                 int M, int N, int K,
                                                 float scale, int mode) {
  const int wave = threadIdx.x >> 5, lane = threadIdx.x & 31;
  const int lrow = lane & 15, hi = lane >> 4;
  const int nBlocks = N / 256;
  const int bm = blockIdx.x / nBlocks, bn = blockIdx.x % nBlocks;
  const int m0 = bm * 64 + (wave & 1) * 32;
  const int n0 = bn * 256 + (wave >> 1) * 64;

  const __bf16* arow0 = A + (size_t)(m0 + lrow) * K + (hi ? 16 : 0);
  const __bf16* arow1 = arow0 + (size_t)16 * K;
  const int koffB = hi ? 16 : 0;
  const __bf16* brow0 = W + (size_t)(n0 + 0  + lrow) * K + koffB;
  const __bf16* brow1 = W + (size_t)(n0 + 16 + lrow) * K + koffB;
  const __bf16* brow2 = W + (size_t)(n0 + 32 + lrow) * K + koffB;
  const __bf16* brow3 = W + (size_t)(n0 + 48 + lrow) * K + koffB;

  v8f acc[2][4] = {{v8f{}, v8f{}, v8f{}, v8f{}}, {v8f{}, v8f{}, v8f{}, v8f{}}};

  // prologue: fragments for k = 0
  v16bf a0 = ldfrag(arow0, arow0 + 8);
  v16bf a1 = ldfrag(arow1, arow1 + 8);
  v16bf b0 = ldfrag(brow0, brow0 + 8);
  v16bf b1 = ldfrag(brow1, brow1 + 8);
  v16bf b2 = ldfrag(brow2, brow2 + 8);
  v16bf b3 = ldfrag(brow3, brow3 + 8);

  for (int k = 32; k < K; k += 32) {
    // issue next k-step's loads first (stay outstanding during WMMAs)
    v16bf na0 = ldfrag(arow0 + k, arow0 + k + 8);
    v16bf na1 = ldfrag(arow1 + k, arow1 + k + 8);
    v16bf nb0 = ldfrag(brow0 + k, brow0 + k + 8);
    v16bf nb1 = ldfrag(brow1 + k, brow1 + k + 8);
    v16bf nb2 = ldfrag(brow2 + k, brow2 + k + 8);
    v16bf nb3 = ldfrag(brow3 + k, brow3 + k + 8);
    acc[0][0] = wmma_bf16(a0, b0, acc[0][0]);
    acc[0][1] = wmma_bf16(a0, b1, acc[0][1]);
    acc[0][2] = wmma_bf16(a0, b2, acc[0][2]);
    acc[0][3] = wmma_bf16(a0, b3, acc[0][3]);
    acc[1][0] = wmma_bf16(a1, b0, acc[1][0]);
    acc[1][1] = wmma_bf16(a1, b1, acc[1][1]);
    acc[1][2] = wmma_bf16(a1, b2, acc[1][2]);
    acc[1][3] = wmma_bf16(a1, b3, acc[1][3]);
    a0 = na0; a1 = na1; b0 = nb0; b1 = nb1; b2 = nb2; b3 = nb3;
  }
  // epilogue: last k-step
  acc[0][0] = wmma_bf16(a0, b0, acc[0][0]);
  acc[0][1] = wmma_bf16(a0, b1, acc[0][1]);
  acc[0][2] = wmma_bf16(a0, b2, acc[0][2]);
  acc[0][3] = wmma_bf16(a0, b3, acc[0][3]);
  acc[1][0] = wmma_bf16(a1, b0, acc[1][0]);
  acc[1][1] = wmma_bf16(a1, b1, acc[1][1]);
  acc[1][2] = wmma_bf16(a1, b2, acc[1][2]);
  acc[1][3] = wmma_bf16(a1, b3, acc[1][3]);

#pragma unroll
  for (int mw = 0; mw < 2; ++mw) {
#pragma unroll
    for (int nt = 0; nt < 4; ++nt) {
#pragma unroll
      for (int r = 0; r < 8; ++r) {
        const int m = m0 + mw * 16 + r + hi * 8;
        const int n = n0 + nt * 16 + lrow;
        const float v = acc[mw][nt][r] * scale;
        if (mode == 2) {
          ((float*)out)[(size_t)m * N + n] = v;
        } else {
          const int b  = m >> 11, t = m & (T_ - 1);
          const int hh = n >> 6,  dh = n & (DH_ - 1);
          const size_t idx = (mode == 0)
              ? ((size_t)((b * H_ + hh) * T_ + t)) * DH_ + dh
              : ((size_t)((b * H_ + hh) * DH_ + dh)) * T_ + t;
          ((unsigned short*)out)[idx] = f2bf(v);
        }
      }
    }
  }
}

// ---------------------------------------------------------------------------
// Stage 3: flash attention. One wave per 16-query tile; 32 keys / iteration.
// Per iteration: issue K-frag loads, then V-frag loads (stay outstanding
// behind softmax), prefetch next tile (global_prefetch_b8), 4 S-WMMAs,
// online softmax (shfl_xor within 16-lane halves), P C->A transpose via LDS,
// 4 PV-WMMAs.  Q pre-scaled by 1/sqrt(DH).
// ---------------------------------------------------------------------------
__global__ __launch_bounds__(256) void flash_attn(const __bf16* __restrict__ Q,   // [B,H,T,64]
                                                  const __bf16* __restrict__ Km,  // [B,H,T,64]
                                                  const __bf16* __restrict__ Vt,  // [B,H,64,T]
                                                  const int*    __restrict__ mask,// [B,T] (nonzero = padded)
                                                  unsigned short* __restrict__ ctx) { // [B,T,H,64] bf16
  __shared__ unsigned short lds_p[8][16][32];          // per-wave 16x32 P tile (8 KB)

  const int wave = threadIdx.x >> 5, lane = threadIdx.x & 31;
  const int lrow = lane & 15, hi = lane >> 4;
  const int bh = blockIdx.x >> 4;                      // 16 query-blocks per (b,h)
  const int tb = blockIdx.x & 15;
  const int b = bh / H_, head = bh % H_;
  const int q0 = (tb * 8 + wave) * 16;

  const __bf16* Qp = Q  + (size_t)bh * T_ * DH_;
  const __bf16* Kp = Km + (size_t)bh * T_ * DH_;
  const __bf16* Vp = Vt + (size_t)bh * DH_ * T_;
  const int*    mp = mask + b * T_;

  // Q A-fragments (K-dim 64 = two 16x32 frags)
  const __bf16* qrow = Qp + (size_t)(q0 + lrow) * DH_ + (hi ? 16 : 0);
  const v16bf qa0 = ldfrag(qrow,      qrow + 8);
  const v16bf qa1 = ldfrag(qrow + 32, qrow + 40);

  v8f o0 = {}, o1 = {}, o2 = {}, o3 = {};
  float mrun[8], lrun[8];
#pragma unroll
  for (int r = 0; r < 8; ++r) { mrun[r] = -1e30f; lrun[r] = 0.0f; }

  const int koffB = hi ? 16 : 0;

  for (int kt = 0; kt < T_; kt += 32) {
    // ---- issue K-fragment loads (keys kt..kt+31, Kdim 64 = 2x2 frags)
    const __bf16* kr0 = Kp + (size_t)(kt + lrow) * DH_ + koffB;
    const __bf16* kr1 = Kp + (size_t)(kt + 16 + lrow) * DH_ + koffB;
    v16bf kb00 = ldfrag(kr0,      kr0 + 8);
    v16bf kb01 = ldfrag(kr0 + 32, kr0 + 40);
    v16bf kb10 = ldfrag(kr1,      kr1 + 8);
    v16bf kb11 = ldfrag(kr1 + 32, kr1 + 40);
    // ---- issue V-fragment loads AFTER K: they stay outstanding through the
    //      softmax (s_wait_loadcnt for the S-WMMAs only covers the K loads)
    const __bf16* vr0 = Vp + (size_t)(lrow)      * T_ + kt + koffB;
    const __bf16* vr1 = Vp + (size_t)(lrow + 16) * T_ + kt + koffB;
    const __bf16* vr2 = Vp + (size_t)(lrow + 32) * T_ + kt + koffB;
    const __bf16* vr3 = Vp + (size_t)(lrow + 48) * T_ + kt + koffB;
    v16bf vb0 = ldfrag(vr0, vr0 + 8);
    v16bf vb1 = ldfrag(vr1, vr1 + 8);
    v16bf vb2 = ldfrag(vr2, vr2 + 8);
    v16bf vb3 = ldfrag(vr3, vr3 + 8);
    const int km0 = mp[kt + lrow];
    const int km1 = mp[kt + 16 + lrow];
    // ---- prefetch next key/value tile into cache (global_prefetch_b8)
    if (kt + 32 < T_) {
      __builtin_prefetch(Kp + (size_t)(kt + 32 + lane) * DH_, 0, 1);
      __builtin_prefetch(Vp + (size_t)(lane << 1) * T_ + kt + 32, 0, 1);
    }
    // ---- S = Q @ K^T : two 16x16 tiles, 4 WMMAs
    v8f s0 = {}, s1 = {};
    s0 = wmma_bf16(qa0, kb00, s0);
    s0 = wmma_bf16(qa1, kb01, s0);
    s1 = wmma_bf16(qa0, kb10, s1);
    s1 = wmma_bf16(qa1, kb11, s1);
    // ---- key-padding mask: column constant per lane in C-layout
#pragma unroll
    for (int r = 0; r < 8; ++r) {
      if (km0) s0[r] = -1e9f;
      if (km1) s1[r] = -1e9f;
    }
    // ---- online softmax (row = r + 8*hi, spread across the 16-lane half)
    float p0[8], p1[8];
#pragma unroll
    for (int r = 0; r < 8; ++r) {
      float v = fmaxf(s0[r], s1[r]);
#pragma unroll
      for (int off = 1; off < 16; off <<= 1) v = fmaxf(v, __shfl_xor(v, off, 32));
      const float mn   = fmaxf(mrun[r], v);
      const float corr = __expf(mrun[r] - mn);
      p0[r] = __expf(s0[r] - mn);
      p1[r] = __expf(s1[r] - mn);
      float rs = p0[r] + p1[r];
#pragma unroll
      for (int off = 1; off < 16; off <<= 1) rs += __shfl_xor(rs, off, 32);
      lrun[r] = lrun[r] * corr + rs;
      mrun[r] = mn;
      o0[r] *= corr; o1[r] *= corr; o2[r] *= corr; o3[r] *= corr;
    }
    // ---- P: C-layout -> A-layout through LDS (uniform loop => barriers OK)
    __syncthreads();
#pragma unroll
    for (int r = 0; r < 8; ++r) {
      lds_p[wave][r + hi * 8][lrow]      = f2bf(p0[r]);
      lds_p[wave][r + hi * 8][lrow + 16] = f2bf(p1[r]);
    }
    __syncthreads();
    const __bf16* prow = (const __bf16*)&lds_p[wave][lrow][hi ? 16 : 0];
    const v16bf pa = ldfrag(prow, prow + 8);
    // ---- O += P @ V : 4 WMMAs (V frags were loaded at the top)
    o0 = wmma_bf16(pa, vb0, o0);
    o1 = wmma_bf16(pa, vb1, o1);
    o2 = wmma_bf16(pa, vb2, o2);
    o3 = wmma_bf16(pa, vb3, o3);
  }

  // ---- normalize, zero padded queries, store ctx [B,T,H,64] bf16
#pragma unroll
  for (int r = 0; r < 8; ++r) {
    const int q   = q0 + r + hi * 8;
    const float z = mp[q] ? 0.0f : (1.0f / lrun[r]);
    const size_t base = ((size_t)(b * T_ + q) * H_ + head) * DH_ + lrow;
    ctx[base +  0] = f2bf(o0[r] * z);
    ctx[base + 16] = f2bf(o1[r] * z);
    ctx[base + 32] = f2bf(o2[r] * z);
    ctx[base + 48] = f2bf(o3[r] * z);
  }
}

// ---------------------------------------------------------------------------
// Launch: cast -> QKV GEMMs -> flash attention -> output GEMM
// ---------------------------------------------------------------------------
extern "C" void kernel_launch(void* const* d_in, const int* in_sizes, int n_in,
                              void* d_out, int out_size, void* d_ws, size_t ws_size,
                              hipStream_t stream) {
  const float* h    = (const float*)d_in[0];
  const int*   mask = (const int*)  d_in[1];
  const float* Wq   = (const float*)d_in[2];
  const float* Wk   = (const float*)d_in[3];
  const float* Wv   = (const float*)d_in[4];
  const float* Wo   = (const float*)d_in[5];

  const int nH = B_ * T_ * D_;        // 6,291,456
  const int nW = D_ * D_;             // 589,824

  char* ws = (char*)d_ws;
  unsigned short* hb  = (unsigned short*)(ws);                           // 12.6 MB
  unsigned short* wqb = (unsigned short*)(ws + 12582912);
  unsigned short* wkb = (unsigned short*)(ws + 12582912 + 1179648);
  unsigned short* wvb = (unsigned short*)(ws + 12582912 + 2 * 1179648);
  unsigned short* wob = (unsigned short*)(ws + 12582912 + 3 * 1179648);
  unsigned short* qb  = (unsigned short*)(ws + 17301504);                // [B,H,T,64]
  unsigned short* kb  = (unsigned short*)(ws + 17301504 + 12582912);     // [B,H,T,64]
  unsigned short* vb  = (unsigned short*)(ws + 17301504 + 2 * 12582912); // [B,H,64,T]
  unsigned short* ctx = (unsigned short*)(ws + 17301504 + 3 * 12582912); // [B,T,H,64]

  cast_f32_bf16<<<nH / 1024, 256, 0, stream>>>(h,  hb,  nH);
  cast_f32_bf16<<<nW / 1024, 256, 0, stream>>>(Wq, wqb, nW);
  cast_f32_bf16<<<nW / 1024, 256, 0, stream>>>(Wk, wkb, nW);
  cast_f32_bf16<<<nW / 1024, 256, 0, stream>>>(Wv, wvb, nW);
  cast_f32_bf16<<<nW / 1024, 256, 0, stream>>>(Wo, wob, nW);

  const int M = B_ * T_;                       // 8192
  const int gemmGrid = (M / 64) * (D_ / 256);  // 128 * 3 = 384 blocks

  // Q scaled by 1/sqrt(DH) = 1/8 at write time (folds softmax scale into Q)
  gemm_bf16<<<gemmGrid, 256, 0, stream>>>((const __bf16*)hb, (const __bf16*)wqb,
                                          qb, M, D_, D_, 0.125f, 0);
  gemm_bf16<<<gemmGrid, 256, 0, stream>>>((const __bf16*)hb, (const __bf16*)wkb,
                                          kb, M, D_, D_, 1.0f, 0);
  gemm_bf16<<<gemmGrid, 256, 0, stream>>>((const __bf16*)hb, (const __bf16*)wvb,
                                          vb, M, D_, D_, 1.0f, 1);

  flash_attn<<<B_ * H_ * (T_ / 128), 256, 0, stream>>>(
      (const __bf16*)qb, (const __bf16*)kb, (const __bf16*)vb, mask, ctx);

  gemm_bf16<<<gemmGrid, 256, 0, stream>>>((const __bf16*)ctx, (const __bf16*)wob,
                                          d_out, M, D_, D_, 1.0f, 2);
}
